// ExtendedFILIP_26946624815126
// MI455X (gfx1250) — matrix operand order repaired and checked by reading, not compiled, not run on hardware
//
#include <hip/hip_runtime.h>
#include <hip/hip_bf16.h>

typedef __attribute__((ext_vector_type(16))) _Float16 v16h;
typedef __attribute__((ext_vector_type(8)))  _Float16 v8h;
typedef __attribute__((ext_vector_type(8)))  float    v8f;
typedef __attribute__((ext_vector_type(4)))  float    v4f;

union F16x16 { v16h v; v8h h[2]; };

#define NEG_HUGE (-3.402823466e38f)

// ---------------------------------------------------------------------------
// Weight transpose + f16 convert:  W [K][N] f32 -> WT [N][K] f16
// ---------------------------------------------------------------------------
__global__ void wtr_k(const float* __restrict__ W, _Float16* __restrict__ WT,
                      int K, int N) {
    int i = blockIdx.x * 256 + threadIdx.x;
    if (i < K * N) {
        int n = i / K;
        int k = i % K;
        WT[i] = (_Float16)W[(size_t)k * N + n];
    }
}

// ---------------------------------------------------------------------------
// GEMM: C[M,N] = A[M,K] (f32) * WT[N,K] (f16, pre-transposed) + bias, opt ReLU
// block 256 = 8 waves; tile 64x64; K-step 32; WMMA f16 -> f32 accum
// ---------------------------------------------------------------------------
__global__ __launch_bounds__(256) void gemm_f16w_k(
    const float* __restrict__ A, const _Float16* __restrict__ WT,
    const float* __restrict__ bias, float* __restrict__ C,
    int M, int N, int K, int relu)
{
    __shared__ _Float16 Alds[64 * 32];
    __shared__ _Float16 Blds[64 * 32];

    const int tid  = threadIdx.x;
    const int lane = tid & 31;
    const int w    = tid >> 5;          // wave 0..7
    const int wm   = w & 3;             // row-tile 0..3  (16 rows each)
    const int wn   = w >> 2;            // col-tile 0..1  (32 cols each)
    const int ln15 = lane & 15;
    const int hi   = lane >> 4;         // 0/1 half of wave
    const int m0   = blockIdx.x * 64;
    const int n0   = blockIdx.y * 64;

    const int arow = tid >> 2;          // 0..63
    const int acol = (tid & 3) * 8;     // 0,8,16,24

    v8f c0 = {}; v8f c1 = {};

    for (int k0 = 0; k0 < K; k0 += 32) {
        // A tile 64x32 f32 -> f16 LDS
        const float* ap = A + (size_t)(m0 + arow) * K + k0 + acol;
        v4f a0 = *(const v4f*)ap;
        v4f a1 = *(const v4f*)(ap + 4);
        v8h ah;
        ah[0] = (_Float16)a0[0]; ah[1] = (_Float16)a0[1];
        ah[2] = (_Float16)a0[2]; ah[3] = (_Float16)a0[3];
        ah[4] = (_Float16)a1[0]; ah[5] = (_Float16)a1[1];
        ah[6] = (_Float16)a1[2]; ah[7] = (_Float16)a1[3];
        *(v8h*)(&Alds[arow * 32 + acol]) = ah;
        // WT tile 64x32 f16 LDS (row = output column, contiguous K)
        *(v8h*)(&Blds[arow * 32 + acol]) =
            *(const v8h*)(WT + (size_t)(n0 + arow) * K + k0 + acol);
        __syncthreads();

        F16x16 af;
        const _Float16* ar = &Alds[(wm * 16 + ln15) * 32];
        af.h[0] = *(const v8h*)(ar + hi * 16);
        af.h[1] = *(const v8h*)(ar + 8 + hi * 16);
        v16h bf0 = *(const v16h*)(&Blds[(wn * 32 + ln15) * 32 + hi * 16]);
        v16h bf1 = *(const v16h*)(&Blds[(wn * 32 + 16 + ln15) * 32 + hi * 16]);

        c0 = __builtin_amdgcn_wmma_f32_16x16x32_f16(false, af.v, false, bf0,
                                                    (short)0, c0, false, false);
        c1 = __builtin_amdgcn_wmma_f32_16x16x32_f16(false, af.v, false, bf1,
                                                    (short)0, c1, false, false);
        __syncthreads();
    }

    const int ncol0 = n0 + wn * 32 + ln15;
    const int ncol1 = ncol0 + 16;
    const float b0 = bias[ncol0];
    const float b1 = bias[ncol1];
    const int mrow = m0 + wm * 16 + hi * 8;
    #pragma unroll
    for (int r = 0; r < 8; ++r) {
        float v0 = c0[r] + b0;
        float v1 = c1[r] + b1;
        if (relu) { v0 = fmaxf(v0, 0.0f); v1 = fmaxf(v1, 0.0f); }
        C[(size_t)(mrow + r) * N + ncol0] = v0;
        C[(size_t)(mrow + r) * N + ncol1] = v1;
    }
}

// ---------------------------------------------------------------------------
// LayerNorm over 512-wide rows (f32 in/out)
// ---------------------------------------------------------------------------
__global__ __launch_bounds__(256) void ln_k(const float* __restrict__ X,
                                            const float* __restrict__ g,
                                            const float* __restrict__ bta,
                                            float* __restrict__ Y) {
    __shared__ float red[256];
    const int row = blockIdx.x, tid = threadIdx.x;
    const float* x = X + (size_t)row * 512;
    float e0 = x[tid], e1 = x[tid + 256];

    red[tid] = e0 + e1; __syncthreads();
    for (int off = 128; off > 0; off >>= 1) {
        if (tid < off) red[tid] += red[tid + off];
        __syncthreads();
    }
    const float mu = red[0] * (1.0f / 512.0f);
    __syncthreads();

    const float d0 = e0 - mu, d1 = e1 - mu;
    red[tid] = d0 * d0 + d1 * d1; __syncthreads();
    for (int off = 128; off > 0; off >>= 1) {
        if (tid < off) red[tid] += red[tid + off];
        __syncthreads();
    }
    const float var = red[0] * (1.0f / 512.0f);
    const float rs = rsqrtf(var + 1e-5f);

    float* y = Y + (size_t)row * 512;
    y[tid]       = d0 * rs * g[tid]       + bta[tid];
    y[tid + 256] = d1 * rs * g[tid + 256] + bta[tid + 256];
}

// ---------------------------------------------------------------------------
// L2-normalize 512-wide rows, emit f16
// ---------------------------------------------------------------------------
__global__ __launch_bounds__(256) void l2h_k(const float* __restrict__ X,
                                             _Float16* __restrict__ Y) {
    __shared__ float red[256];
    const int row = blockIdx.x, tid = threadIdx.x;
    const float* x = X + (size_t)row * 512;
    float e0 = x[tid], e1 = x[tid + 256];
    red[tid] = e0 * e0 + e1 * e1; __syncthreads();
    for (int off = 128; off > 0; off >>= 1) {
        if (tid < off) red[tid] += red[tid + off];
        __syncthreads();
    }
    const float rs = rsqrtf(red[0]);
    _Float16* y = Y + (size_t)row * 512;
    y[tid]       = (_Float16)(e0 * rs);
    y[tid + 256] = (_Float16)(e1 * rs);
}

// ---------------------------------------------------------------------------
// Similarity + masked max/mean. One block per (a,b) pair; grid (32,32).
// 8 waves; wave w owns rows t = w*16..w*16+15 (A fragments resident in VGPRs).
// B streamed via double-buffered LDS using gfx1250 async global->LDS copies
// (GLOBAL_LOAD_ASYNC_TO_LDS_B128 + s_wait_asynccnt), overlapped with WMMA.
// ---------------------------------------------------------------------------
__global__ __launch_bounds__(256) void sim_k(
    const _Float16* __restrict__ HA, const _Float16* __restrict__ HB,
    const int* __restrict__ maskA, const int* __restrict__ maskB,
    const float* __restrict__ temp, float* __restrict__ out)
{
    __shared__ _Float16 Blds[2][16 * 512];   // 2 x 16 KB double buffer
    __shared__ float sB[1024];
    __shared__ float sA[128];
    __shared__ float colMax[8][16];
    __shared__ int   mA[128];
    __shared__ int   mB[1024];
    __shared__ float redS[256];
    __shared__ float redC[256];

    const int a = blockIdx.x, b = blockIdx.y;
    const int tid  = threadIdx.x;
    const int lane = tid & 31;
    const int w    = tid >> 5;
    const int ln15 = lane & 15;
    const int hi   = lane >> 4;
    const float invT = 1.0f / temp[0];

    if (tid < 128) mA[tid] = maskA[a * 128 + tid];
    for (int i = tid; i < 1024; i += 256) mB[i] = maskB[b * 1024 + i];

    // async staging coords: each thread moves 64 contiguous bytes per chunk
    const int bc   = tid >> 4;            // 0..15 : column (row of HB) in chunk
    const int boff = (tid & 15) * 32;     // halves (64 B granule)
    const unsigned long long gaBase =
        (unsigned long long)(uintptr_t)(HB + (size_t)(b * 1024 + bc) * 512 + boff);
    const unsigned lds0 = (unsigned)(uintptr_t)&Blds[0][bc * 512 + boff];
    const unsigned lds1 = (unsigned)(uintptr_t)&Blds[1][bc * 512 + boff];

    // kick off chunk 0 -> Blds[0]
    asm volatile(
        "global_load_async_to_lds_b128 %0, %1, off\n\t"
        "global_load_async_to_lds_b128 %0, %1, off offset:16\n\t"
        "global_load_async_to_lds_b128 %0, %1, off offset:32\n\t"
        "global_load_async_to_lds_b128 %0, %1, off offset:48"
        :: "v"(lds0), "v"(gaBase) : "memory");

    // preload this wave's A fragments: rows a*128 + w*16 + ln15, all K
    F16x16 af[16];
    const _Float16* ar = HA + (size_t)(a * 128 + w * 16 + ln15) * 512;
    #pragma unroll
    for (int k = 0; k < 16; ++k) {
        af[k].h[0] = *(const v8h*)(ar + k * 32 + hi * 16);
        af[k].h[1] = *(const v8h*)(ar + k * 32 + 8 + hi * 16);
    }
    __syncthreads();   // masks ready

    int amask[8];
    float runA[8];
    #pragma unroll
    for (int r = 0; r < 8; ++r) {
        amask[r] = mA[w * 16 + hi * 8 + r];
        runA[r]  = NEG_HUGE;
    }

    for (int ch = 0; ch < 64; ++ch) {
        const int cur = ch & 1;
        // current chunk's async writes complete (own wave), publish to WGP
        asm volatile("s_wait_asynccnt 0x0" ::: "memory");
        __syncthreads();

        // issue next chunk into the other buffer; overlaps with WMMA below
        if (ch + 1 < 64) {
            const unsigned long long ga =
                gaBase + (unsigned long long)(ch + 1) * 16384ull; // 16 rows * 1 KB
            const unsigned ldst = ((ch + 1) & 1) ? lds1 : lds0;
            asm volatile(
                "global_load_async_to_lds_b128 %0, %1, off\n\t"
                "global_load_async_to_lds_b128 %0, %1, off offset:16\n\t"
                "global_load_async_to_lds_b128 %0, %1, off offset:32\n\t"
                "global_load_async_to_lds_b128 %0, %1, off offset:48"
                :: "v"(ldst), "v"(ga) : "memory");
        }

        const int s0 = ch * 16;
        v8f c0 = {};
        const _Float16* bp = &Blds[cur][ln15 * 512 + hi * 16];
        #pragma unroll
        for (int k = 0; k < 16; ++k) {
            v16h bf = *(const v16h*)(bp + k * 32);
            c0 = __builtin_amdgcn_wmma_f32_16x16x32_f16(false, af[k].v, false, bf,
                                                        (short)0, c0, false, false);
        }

        // masked reductions for this chunk
        const int bm = mB[s0 + ln15];
        float cm = NEG_HUGE;
        #pragma unroll
        for (int r = 0; r < 8; ++r) {
            float v0 = (amask[r] && bm) ? c0[r] * invT : NEG_HUGE;
            runA[r] = fmaxf(runA[r], v0);
            cm = fmaxf(cm, v0);
        }
        cm = fmaxf(cm, __shfl_xor(cm, 16, 32));
        if (hi == 0) colMax[w][ln15] = cm;
        __syncthreads();
        if (tid < 16) {
            float m = colMax[0][tid];
            #pragma unroll
            for (int ww = 1; ww < 8; ++ww) m = fmaxf(m, colMax[ww][tid]);
            sB[s0 + tid] = m;   // each s written exactly once
        }
        // next iteration's top barrier protects colMax reuse and buffer swap
    }
    __syncthreads();

    // sA: reduce runA across the 16 lanes of each half-wave
    #pragma unroll
    for (int r = 0; r < 8; ++r) {
        float v = runA[r];
        v = fmaxf(v, __shfl_xor(v, 1, 32));
        v = fmaxf(v, __shfl_xor(v, 2, 32));
        v = fmaxf(v, __shfl_xor(v, 4, 32));
        v = fmaxf(v, __shfl_xor(v, 8, 32));
        if (ln15 == 0) sA[w * 16 + hi * 8 + r] = v;
    }
    __syncthreads();

    // masked mean over t (128)
    float vs = 0.0f, vc = 0.0f;
    if (tid < 128 && mA[tid]) { vs = sA[tid]; vc = 1.0f; }
    redS[tid] = vs; redC[tid] = vc;
    __syncthreads();
    for (int off = 128; off > 0; off >>= 1) {
        if (tid < off) { redS[tid] += redS[tid + off]; redC[tid] += redC[tid + off]; }
        __syncthreads();
    }
    float scoreA = 0.0f;
    if (tid == 0) scoreA = redS[0] / fmaxf(redC[0], 1e-6f);
    __syncthreads();

    // masked mean over s (1024)
    vs = 0.0f; vc = 0.0f;
    for (int i = tid; i < 1024; i += 256)
        if (mB[i]) { vs += sB[i]; vc += 1.0f; }
    redS[tid] = vs; redC[tid] = vc;
    __syncthreads();
    for (int off = 128; off > 0; off >>= 1) {
        if (tid < off) { redS[tid] += redS[tid + off]; redC[tid] += redC[tid + off]; }
        __syncthreads();
    }
    if (tid == 0) {
        out[a * 32 + b]        = scoreA;
        out[1024 + a * 32 + b] = redS[0] / fmaxf(redC[0], 1e-6f);
    }
}

// ---------------------------------------------------------------------------
// Host launcher
// ---------------------------------------------------------------------------
extern "C" void kernel_launch(void* const* d_in, const int* in_sizes, int n_in,
                              void* d_out, int out_size, void* d_ws, size_t ws_size,
                              hipStream_t stream) {
    (void)in_sizes; (void)n_in; (void)out_size; (void)ws_size;

    constexpr int Bn = 32, TA = 128, TB = 1024, DIN = 1280, DEMB = 512;
    constexpr int MA = Bn * TA;     // 4096
    constexpr int MB = Bn * TB;     // 32768

    const float* pep_esm  = (const float*)d_in[0];
    const float* rec_esm  = (const float*)d_in[1];
    const int*   pep_mask = (const int*)d_in[2];
    const int*   rec_mask = (const int*)d_in[3];
    const float* temp     = (const float*)d_in[4];
    const float* pep_pw  = (const float*)d_in[5];
    const float* pep_pb  = (const float*)d_in[6];
    const float* pep_f1w = (const float*)d_in[7];
    const float* pep_f1b = (const float*)d_in[8];
    const float* pep_lng = (const float*)d_in[9];
    const float* pep_lnb = (const float*)d_in[10];
    const float* pep_f2w = (const float*)d_in[11];
    const float* pep_f2b = (const float*)d_in[12];
    const float* rec_pw  = (const float*)d_in[13];
    const float* rec_pb  = (const float*)d_in[14];
    const float* rec_f1w = (const float*)d_in[15];
    const float* rec_f1b = (const float*)d_in[16];
    const float* rec_lng = (const float*)d_in[17];
    const float* rec_lnb = (const float*)d_in[18];
    const float* rec_f2w = (const float*)d_in[19];
    const float* rec_f2b = (const float*)d_in[20];

    char* ws = (char*)d_ws;
    _Float16* wt_pep_p  = (_Float16*)(ws);                               // 512*1280
    _Float16* wt_pep_f1 = (_Float16*)(ws + 1310720);                     // 512*512
    _Float16* wt_pep_f2 = (_Float16*)(ws + 1310720 + 524288);
    _Float16* wt_rec_p  = (_Float16*)(ws + 2359296);
    _Float16* wt_rec_f1 = (_Float16*)(ws + 2359296 + 1310720);
    _Float16* wt_rec_f2 = (_Float16*)(ws + 2359296 + 1310720 + 524288);
    _Float16* hA16 = (_Float16*)(ws + 4718592);                          // 4096*512
    _Float16* hB16 = (_Float16*)(ws + 4718592 + 4194304);                // 32768*512
    float* buf0 = (float*)(ws + 42467328);                               // 32768*512 f32
    float* buf1 = (float*)(ws + 42467328 + 67108864);

    // 1) weight transposes
    wtr_k<<<(DIN * DEMB + 255) / 256, 256, 0, stream>>>(pep_pw, wt_pep_p, DIN, DEMB);
    wtr_k<<<(DEMB * DEMB + 255) / 256, 256, 0, stream>>>(pep_f1w, wt_pep_f1, DEMB, DEMB);
    wtr_k<<<(DEMB * DEMB + 255) / 256, 256, 0, stream>>>(pep_f2w, wt_pep_f2, DEMB, DEMB);
    wtr_k<<<(DIN * DEMB + 255) / 256, 256, 0, stream>>>(rec_pw, wt_rec_p, DIN, DEMB);
    wtr_k<<<(DEMB * DEMB + 255) / 256, 256, 0, stream>>>(rec_f1w, wt_rec_f1, DEMB, DEMB);
    wtr_k<<<(DEMB * DEMB + 255) / 256, 256, 0, stream>>>(rec_f2w, wt_rec_f2, DEMB, DEMB);

    // 2) pep encode
    {
        dim3 g1(MA / 64, DEMB / 64);
        gemm_f16w_k<<<g1, 256, 0, stream>>>(pep_esm, wt_pep_p, pep_pb, buf0, MA, DEMB, DIN, 0);
        gemm_f16w_k<<<g1, 256, 0, stream>>>(buf0, wt_pep_f1, pep_f1b, buf1, MA, DEMB, DEMB, 1);
        ln_k<<<MA, 256, 0, stream>>>(buf1, pep_lng, pep_lnb, buf0);
        gemm_f16w_k<<<g1, 256, 0, stream>>>(buf0, wt_pep_f2, pep_f2b, buf1, MA, DEMB, DEMB, 0);
        l2h_k<<<MA, 256, 0, stream>>>(buf1, hA16);
    }
    // 3) rec encode
    {
        dim3 g1(MB / 64, DEMB / 64);
        gemm_f16w_k<<<g1, 256, 0, stream>>>(rec_esm, wt_rec_p, rec_pb, buf0, MB, DEMB, DIN, 0);
        gemm_f16w_k<<<g1, 256, 0, stream>>>(buf0, wt_rec_f1, rec_f1b, buf1, MB, DEMB, DEMB, 1);
        ln_k<<<MB, 256, 0, stream>>>(buf1, rec_lng, rec_lnb, buf0);
        gemm_f16w_k<<<g1, 256, 0, stream>>>(buf0, wt_rec_f2, rec_f2b, buf1, MB, DEMB, DEMB, 0);
        l2h_k<<<MB, 256, 0, stream>>>(buf1, hB16);
    }
    // 4) similarity + masked max/mean
    sim_k<<<dim3(32, 32), 256, 0, stream>>>(hA16, hB16, pep_mask, rec_mask, temp,
                                            (float*)d_out);
}
